// Correlation_network_21165598834698
// MI455X (gfx1250) — matrix operand order repaired and checked by reading, not compiled, not run on hardware
//
#include <hip/hip_runtime.h>
#include <hip/hip_bf16.h>
#include <stdint.h>

// ---- WMMA vector types (CDNA5 / gfx1250, wave32) ----
typedef __attribute__((ext_vector_type(16))) __bf16 v16bf;
typedef __attribute__((ext_vector_type(8)))  float  v8f;

#define TILE    128     // output tile edge per workgroup
#define KCHUNK  128     // K staged per LDS round (K=256 -> 2 rounds)
#define LDSROW  136     // bf16 elems per LDS row: 128 + 8 pad (272B = 68 dwords, bank-skewed)
#define MDIM    2304    // 48*48
#define KDIM    256

union FragU { v16bf v; uint4 q[2]; };
union Pack4 { __bf16 h[4]; uint2 u; };

__global__ __launch_bounds__(256)
void corr_volume_wmma_bf16(const float* __restrict__ A,
                           const float* __restrict__ B,
                           float* __restrict__ Out)
{
    extern __shared__ __bf16 lds[];
    __bf16* Al = lds;                    // [TILE][LDSROW]
    __bf16* Bl = lds + TILE * LDSROW;    // [TILE][LDSROW]

    const int bt = blockIdx.z;
    const int mb = blockIdx.y * TILE;    // row-tile base (i*W+j of A)
    const int nb = blockIdx.x * TILE;    // col-tile base (k*W+l of B)

    const float* Ab = A   + (size_t)bt * MDIM * KDIM;
    const float* Bb = B   + (size_t)bt * MDIM * KDIM;
    float*       Ob = Out + (size_t)bt * MDIM * MDIM;

    const int tid  = threadIdx.x;
    const int lane = tid & 31;
    const int wave = tid >> 5;
    const int wm   = wave >> 1;          // 0..3 : wave's 32-row band
    const int wn   = wave & 1;           // 0..1 : wave's 64-col band
    const int half = lane >> 4;          // 0/1 : lane half (K/M split per ISA layout)
    const int row16 = lane & 15;

    // LDS base pointers for this lane's fragment rows
    const __bf16* ArowBase = Al + (wm * 32 + row16) * LDSROW + half * 16;
    const __bf16* BrowBase = Bl + (wn * 64 + row16) * LDSROW + half * 16;

    v8f acc[2][4];
#pragma unroll
    for (int tm = 0; tm < 2; ++tm)
#pragma unroll
        for (int tn = 0; tn < 4; ++tn)
#pragma unroll
            for (int e = 0; e < 8; ++e) acc[tm][tn][e] = 0.0f;

    for (int kc = 0; kc < KDIM; kc += KCHUNK) {
        // ---- stage A & B chunk: 128 rows x 128 f32 -> bf16 in LDS ----
        // 4096 float4 per matrix / 256 threads = 16 each; consecutive tids
        // read consecutive float4 (512B/warp, fully coalesced).
#pragma unroll
        for (int it = 0; it < 16; ++it) {
            const int q   = it * 256 + tid;     // 0..4095
            const int row = q >> 5;             // 32 float4 per row
            const int c4  = (q & 31) << 2;      // col in floats
            const float4 av = *(const float4*)(Ab + (size_t)(mb + row) * KDIM + kc + c4);
            const float4 bv = *(const float4*)(Bb + (size_t)(nb + row) * KDIM + kc + c4);
            Pack4 pa, pb;
            pa.h[0] = (__bf16)av.x; pa.h[1] = (__bf16)av.y;
            pa.h[2] = (__bf16)av.z; pa.h[3] = (__bf16)av.w;
            pb.h[0] = (__bf16)bv.x; pb.h[1] = (__bf16)bv.y;
            pb.h[2] = (__bf16)bv.z; pb.h[3] = (__bf16)bv.w;
            *(uint2*)(Al + row * LDSROW + c4) = pa.u;   // ds_store_b64
            *(uint2*)(Bl + row * LDSROW + c4) = pb.u;   // ds_store_b64
        }
        __syncthreads();

        // ---- compute: 4 K-steps of 32 per chunk, B-fragments pipelined ----
#pragma unroll
        for (int ks = 0; ks < KCHUNK / 32; ++ks) {
            // A fragments (16x32 bf16): lane half h of row m holds
            // K = h*16 .. h*16+15 -> 32 contiguous bytes at (ks*64 + h*32)
            FragU af[2];
#pragma unroll
            for (int tm = 0; tm < 2; ++tm) {
                const __bf16* rp = ArowBase + tm * (16 * LDSROW) + ks * 32;
                af[tm].q[0] = *(const uint4*)(rp);        // ds_load_b128
                af[tm].q[1] = *(const uint4*)(rp + 8);    // ds_load_b128
            }
            // B fragments (32x16 bf16, K x N): lane half h of col n holds
            // K = h*16 .. h*16+15 -> 32 contiguous bytes at (ks*64 + h*32).
            // Pipeline one fragment ahead so WMMA never hard-waits on dscnt==0.
            FragU bcur, bnxt;
            {
                const __bf16* rp = BrowBase + ks * 32;
                bcur.q[0] = *(const uint4*)(rp);
                bcur.q[1] = *(const uint4*)(rp + 8);
            }
#pragma unroll
            for (int tn = 0; tn < 4; ++tn) {
                if (tn < 3) {
                    const __bf16* rp = BrowBase + (tn + 1) * (16 * LDSROW) + ks * 32;
                    bnxt.q[0] = *(const uint4*)(rp);      // prefetch next B frag
                    bnxt.q[1] = *(const uint4*)(rp + 8);
                }
#pragma unroll
                for (int tm = 0; tm < 2; ++tm)
                    acc[tm][tn] = __builtin_amdgcn_wmma_f32_16x16x32_bf16(
                        /*neg_a=*/false, af[tm].v,
                        /*neg_b=*/false, bcur.v,
                        /*c_mod=*/(short)0, acc[tm][tn],
                        /*reuse_a=*/false, /*reuse_b=*/false);
                bcur = bnxt;
            }
        }
        __syncthreads();
    }

    // ---- store: C/D layout (ISA 7.12.2): VGPR v -> M = half*8 + v, N = lane&15 ----
    // Output is write-once / never re-read (170MB > useful L2 share), while A/B
    // tiles are re-read 18x and fit in L2: stream results with TH=NT.
#pragma unroll
    for (int tm = 0; tm < 2; ++tm)
#pragma unroll
        for (int tn = 0; tn < 4; ++tn) {
            const int col = nb + wn * 64 + tn * 16 + row16;
#pragma unroll
            for (int v = 0; v < 8; ++v) {
                const int row = mb + wm * 32 + tm * 16 + half * 8 + v;
                __builtin_nontemporal_store(acc[tm][tn][v],
                                            Ob + (size_t)row * MDIM + col);
            }
        }
}

extern "C" void kernel_launch(void* const* d_in, const int* in_sizes, int n_in,
                              void* d_out, int out_size, void* d_ws, size_t ws_size,
                              hipStream_t stream) {
    (void)in_sizes; (void)n_in; (void)d_ws; (void)ws_size; (void)out_size;
    const float* A = (const float*)d_in[0];
    const float* B = (const float*)d_in[1];
    float* Out = (float*)d_out;

    dim3 grid(MDIM / TILE, MDIM / TILE, 8);   // (18, 18, 8)
    dim3 block(256);
    const size_t ldsBytes = (size_t)2 * TILE * LDSROW * sizeof(__bf16); // 69632 B
    corr_volume_wmma_bf16<<<grid, block, ldsBytes, stream>>>(A, B, Out);
}
